// Memory_11879879543888
// MI455X (gfx1250) — matrix operand order repaired and checked by reading, not compiled, not run on hardware
//
#include <hip/hip_runtime.h>

// ---------------- CDNA5 WMMA types ----------------
typedef __bf16 v16bf __attribute__((ext_vector_type(16)));
typedef float  v8f   __attribute__((ext_vector_type(8)));

// Problem constants (from reference): B=4, D_e=128, D_o=512, T=8, H=W=32
#define NB  4
#define DE  128
#define DO  512
#define NN  8192    // T*H*W memory keys
#define MM  1024    // H*W query positions

// fp32 -> bf16 round-to-nearest-even
__device__ __forceinline__ unsigned short f2bf(float f) {
  union { float f; unsigned int u; } c; c.f = f;
  unsigned int u = c.u;
  u += 0x7fffu + ((u >> 16) & 1u);
  return (unsigned short)(u >> 16);
}

// Fragment index helpers, contiguous k = 16*(lane/16) + elem (16-bit A 16x32, B 32x16, 32-bit C 16x16)
__device__ __forceinline__ int laneA(int r, int k) { return r + ((k & 16) ? 16 : 0); }
__device__ __forceinline__ int elemA(int k)        { return k & 15; }
__device__ __forceinline__ int laneB(int c, int k) { return c + ((k & 16) ? 16 : 0); }

// ============================================================
// Kernel 1: S[b][n][m] = (1/sqrt(128)) * sum_d mi[b][d][n]*qi[b][d][m]
// tile 128(n) x 128(m), K-step 32 (4 steps), double-buffered LDS,
// float4 staging: 1 global_load_b128 per 4 elements.
// ============================================================
__global__ __launch_bounds__(256)
void k_scores(const float* __restrict__ mi, const float* __restrict__ qi,
              float* __restrict__ S) {
  const int t  = threadIdx.x;
  const int m0 = blockIdx.x * 128;
  const int n0 = blockIdx.y * 128;
  const int b  = blockIdx.z;
  const int w    = t >> 5;
  const int lane = t & 31;

  __shared__ __align__(32) unsigned short lA[2][8 * 32 * 16];
  __shared__ __align__(32) unsigned short lB[2][8 * 32 * 16];

  const float* miB = mi + (size_t)b * DE * NN;
  const float* qiB = qi + (size_t)b * DE * MM;

  v8f acc[8] = {};

  // ---- staging: tiles viewed as [32 d][128 x], x contiguous in memory ----
  auto stage = [&](int buf, int kk) {
#pragma unroll
    for (int i = 0; i < 4; ++i) {
      const int idx4 = t + i * 256;        // 0..1023 float4s
      const int d  = idx4 >> 5;            // 0..31
      const int x4 = (idx4 & 31) * 4;      // 0..124
      const int nb = x4 >> 4;
      const int r0 = x4 & 15;              // +3 stays in block
      float4 va = *(const float4*)&miB[(size_t)(kk + d) * NN + n0 + x4];
      unsigned short* pa = &lA[buf][((nb * 32) + laneA(r0, d)) * 16 + elemA(d)];
      pa[0]  = f2bf(va.x); pa[16] = f2bf(va.y);
      pa[32] = f2bf(va.z); pa[48] = f2bf(va.w);
      float4 vb = *(const float4*)&qiB[(size_t)(kk + d) * MM + m0 + x4];
      unsigned short* pb = &lB[buf][((nb * 32) + laneB(r0, d)) * 16 + (d & 15)];
      pb[0]  = f2bf(vb.x); pb[16] = f2bf(vb.y);
      pb[32] = f2bf(vb.z); pb[48] = f2bf(vb.w);
    }
  };

  stage(0, 0);
  __syncthreads();
  for (int s = 0; s < DE / 32; ++s) {
    if (s + 1 < DE / 32) stage((s + 1) & 1, (s + 1) * 32);
    const int p = s & 1;
    v16bf a = *(const v16bf*)&lA[p][(w * 32 + lane) * 16];
#pragma unroll
    for (int mb = 0; mb < 8; ++mb) {
      v16bf bm = *(const v16bf*)&lB[p][(mb * 32 + lane) * 16];
      acc[mb] = __builtin_amdgcn_wmma_f32_16x16x32_bf16(
          false, a, false, bm, (short)0, acc[mb], false, false);
    }
    __syncthreads();
  }

  const float scale = 0.0883883476483184f; // 1/sqrt(128)
  const int rowb = (lane >> 4) * 8;
  const int col  = lane & 15;
  float* Sb = S + (size_t)b * NN * MM;
#pragma unroll
  for (int mb = 0; mb < 8; ++mb) {
#pragma unroll
    for (int v = 0; v < 8; ++v) {
      int n = n0 + w * 16 + rowb + v;
      int m = m0 + mb * 16 + col;
      Sb[(size_t)n * MM + m] = acc[mb][v] * scale;
    }
  }
}

// ============================================================
// Kernel 2: per-column online softmax stats -> (max, 1/sum)
// ============================================================
__global__ __launch_bounds__(256)
void k_stats(const float* __restrict__ S, float2* __restrict__ stats) {
  const int tx = threadIdx.x & 31;
  const int ty = threadIdx.x >> 5;
  const int m  = blockIdx.x * 32 + tx;
  const int b  = blockIdx.y;
  const float* col = S + (size_t)b * NN * MM + m;

  float mx = -3.0e38f, sm = 0.0f;
  for (int n = ty; n < NN; n += 8) {
    float x  = col[(size_t)n * MM];
    float nm = fmaxf(mx, x);
    sm = sm * __expf(mx - nm) + __expf(x - nm);
    mx = nm;
  }
  __shared__ float smx[8][32];
  __shared__ float ssm[8][32];
  smx[ty][tx] = mx; ssm[ty][tx] = sm;
  __syncthreads();
  if (ty == 0) {
#pragma unroll
    for (int j = 1; j < 8; ++j) {
      float m2 = smx[j][tx], s2 = ssm[j][tx];
      float nm = fmaxf(mx, m2);
      sm = sm * __expf(mx - nm) + s2 * __expf(m2 - nm);
      mx = nm;
    }
    stats[(size_t)b * MM + m] = make_float2(mx, 1.0f / sm);
  }
}

// ============================================================
// Kernel 3: in-place P = exp(S - max) * rinv  (float4, S resident in L2)
// ============================================================
__global__ __launch_bounds__(256)
void k_norm(float* __restrict__ S, const float2* __restrict__ stats) {
  size_t i4   = (size_t)blockIdx.x * 256 + threadIdx.x;  // 8,388,608 vec4
  size_t base = i4 * 4;
  int b = (int)(base >> 23);          // N*M = 2^23 per batch
  int m = (int)(base & (MM - 1));     // 4 consecutive m
  const float2* st = stats + (size_t)b * MM + m;
  float4 x = *(const float4*)(S + base);
  float2 s0 = st[0], s1 = st[1], s2 = st[2], s3 = st[3];
  x.x = __expf(x.x - s0.x) * s0.y;
  x.y = __expf(x.y - s1.x) * s1.y;
  x.z = __expf(x.z - s2.x) * s2.y;
  x.w = __expf(x.w - s3.x) * s3.y;
  *(float4*)(S + base) = x;
}

// ============================================================
// Kernel 4: out[b][do][m] = sum_n mo[b][do][n] * P[b][n][m]
// tile 128(do) x 64(m), K=8192 in 256 steps, double-buffered LDS.
// A staging: 4 consecutive K -> 4 consecutive LDS ushorts (one b64 store).
// ============================================================
__global__ __launch_bounds__(256)
void k_gemm2(const float* __restrict__ mo, const float* __restrict__ P,
             float* __restrict__ out) {
  const int t   = threadIdx.x;
  const int m0  = blockIdx.x * 64;
  const int do0 = blockIdx.y * 128;
  const int b   = blockIdx.z;
  const int w    = t >> 5;
  const int lane = t & 31;

  __shared__ __align__(32) unsigned short lA[2][8 * 32 * 16]; // 128x32 bf16
  __shared__ __align__(32) unsigned short lB[2][4 * 32 * 16]; // 32x64 bf16

  const float* moB = mo + (size_t)b * DO * NN;
  const float* PB  = P  + (size_t)b * NN * MM;

  v8f acc[4] = {};

  auto stage = [&](int buf, int kk) {
    // A tile [128 do][32 n], n (K) contiguous: pack 4 bf16 -> one 8B LDS store
#pragma unroll
    for (int i = 0; i < 4; ++i) {
      const int idx4 = t + i * 256;        // 0..1023
      const int dl = idx4 >> 3;            // 0..127
      const int n4 = (idx4 & 7) * 4;       // 0..28
      float4 v = *(const float4*)&moB[(size_t)(do0 + dl) * NN + kk + n4];
      unsigned int u0 = (unsigned int)f2bf(v.x) | ((unsigned int)f2bf(v.y) << 16);
      unsigned int u1 = (unsigned int)f2bf(v.z) | ((unsigned int)f2bf(v.w) << 16);
      unsigned int* pa = (unsigned int*)
          &lA[buf][(((dl >> 4) * 32) + laneA(dl & 15, n4)) * 16 + elemA(n4)];
      pa[0] = u0; pa[1] = u1;
    }
    // B tile [32 n][64 m], m contiguous
#pragma unroll
    for (int i = 0; i < 2; ++i) {
      const int idx4 = t + i * 256;        // 0..511
      const int d  = idx4 >> 4;            // 0..31
      const int m4 = (idx4 & 15) * 4;      // 0..60
      float4 v = *(const float4*)&PB[(size_t)(kk + d) * MM + m0 + m4];
      unsigned short* pb =
          &lB[buf][(((m4 >> 4) * 32) + laneB(m4 & 15, d)) * 16 + (d & 15)];
      pb[0]  = f2bf(v.x); pb[16] = f2bf(v.y);
      pb[32] = f2bf(v.z); pb[48] = f2bf(v.w);
    }
  };

  stage(0, 0);
  __syncthreads();
  for (int s = 0; s < NN / 32; ++s) {
    const int kk = s * 32;
    if (kk + 32 < NN) stage((s + 1) & 1, kk + 32);
    if (kk + 64 < NN) { // prefetch two steps ahead
      __builtin_prefetch(PB + (size_t)(kk + 64 + (t >> 3)) * MM + m0 + (t & 7) * 8, 0, 1);
      __builtin_prefetch(moB + (size_t)(do0 + (t >> 1)) * NN + kk + 64 + (t & 1) * 16, 0, 1);
    }
    const int p = s & 1;
    v16bf a = *(const v16bf*)&lA[p][(w * 32 + lane) * 16];
#pragma unroll
    for (int mb = 0; mb < 4; ++mb) {
      v16bf bm = *(const v16bf*)&lB[p][(mb * 32 + lane) * 16];
      acc[mb] = __builtin_amdgcn_wmma_f32_16x16x32_bf16(
          false, a, false, bm, (short)0, acc[mb], false, false);
    }
    __syncthreads();
  }

  const int rowb = (lane >> 4) * 8;
  const int col  = lane & 15;
  float* ob = out + (size_t)b * (2 * DO) * MM;  // channels 0..511
#pragma unroll
  for (int mb = 0; mb < 4; ++mb) {
#pragma unroll
    for (int v = 0; v < 8; ++v) {
      int d = do0 + w * 16 + rowb + v;
      int m = m0 + mb * 16 + col;
      ob[(size_t)d * MM + m] = acc[mb][v];
    }
  }
}

// ============================================================
// Kernel 5: out[b][512+c][m] = query_out[b][c][m]  (float4 copy)
// ============================================================
__global__ __launch_bounds__(256)
void k_copy(const float4* __restrict__ qo, float4* __restrict__ out) {
  size_t i4 = (size_t)blockIdx.x * 256 + threadIdx.x;  // 524,288 vec4 total
  int b = (int)(i4 >> 17);            // 131072 vec4 per batch
  size_t off = i4 & 131071u;
  out[(size_t)b * 262144 + 131072 + off] = qo[(size_t)b * 131072 + off];
}

extern "C" void kernel_launch(void* const* d_in, const int* in_sizes, int n_in,
                              void* d_out, int out_size, void* d_ws, size_t ws_size,
                              hipStream_t stream) {
  (void)in_sizes; (void)n_in; (void)out_size; (void)ws_size;
  const float* mi = (const float*)d_in[0];  // memory_in  [4][128][8192]
  const float* mo = (const float*)d_in[1];  // memory_out [4][512][8192]
  const float* qi = (const float*)d_in[2];  // query_in   [4][128][1024]
  const float* qo = (const float*)d_in[3];  // query_out  [4][512][1024]

  float* out  = (float*)d_out;                          // [4][1024][1024]
  float* prob = out + (size_t)NB * (2 * DO) * MM;       // [4][8192][1024]
  float2* stats = (float2*)d_ws;                        // [4][1024] (max, 1/sum)

  // 1) scores -> prob region (unnormalized)
  k_scores<<<dim3(MM / 128, NN / 128, NB), 256, 0, stream>>>(mi, qi, prob);
  // 2) per-column online softmax stats
  k_stats<<<dim3(MM / 32, NB), 256, 0, stream>>>(prob, stats);
  // 3) normalize in place (L2-resident)
  k_norm<<<(NB * NN * MM) / (4 * 256), 256, 0, stream>>>(prob, stats);
  // 4) memory = mo @ prob -> out[:, 0:512]
  k_gemm2<<<dim3(MM / 64, DO / 128, NB), 256, 0, stream>>>(mo, prob, out);
  // 5) out[:, 512:1024] = query_out
  k_copy<<<(NB * DO * MM) / (4 * 256), 256, 0, stream>>>((const float4*)qo, (float4*)out);
}
